// CausalMultiHypothesisTransformerLayer_3204045603765
// MI455X (gfx1250) — matrix operand order repaired and checked by reading, not compiled, not run on hardware
//
#include <hip/hip_runtime.h>
#include <cstdint>

typedef __attribute__((ext_vector_type(16))) _Float16 v16h;
typedef __attribute__((ext_vector_type(8)))  float    v8f;
typedef __attribute__((ext_vector_type(4)))  unsigned v4u;
typedef __attribute__((ext_vector_type(8)))  int      v8i;
typedef __attribute__((ext_vector_type(4)))  int      v4i;

#define NTOK 4096   // S*B = 1024*4 tokens
#define EDIM 256
#define FDIM 512

// ---------------------------------------------------------------------------
// Tiled WMMA GEMM:  out = act(A(f16)[N,K] @ W(f32) + bias) (+residual)
//   BT=false: W stored [K,M] (jax [in,out]);  BT=true: W stored [M,K] (torch)
// Weight tile is staged into LDS by the Tensor Data Mover (TDM), then
// converted f32->f16 into a fragment-friendly transposed layout.
// Block: 256 threads = 8 waves; block tile 64(rows) x 128(cols); wave 32x32.
// ---------------------------------------------------------------------------
template<bool BT, bool RELU, bool HB, bool HR, bool OF, bool OH>
__global__ __launch_bounds__(256) void gemm_f16_wmma(
    const _Float16* __restrict__ A,
    const float* __restrict__ W,
    const float* __restrict__ bias,
    const float* __restrict__ resid,
    float* __restrict__ outF,
    _Float16* __restrict__ outH,
    int K, int M)
{
  __shared__ _Float16 lA[64][40];     // A tile, row-major [row][k]
  __shared__ _Float16 lBt[128][40];   // B tile transposed: [n][k] (fragment order)
  __shared__ float    lWraw[32 * 128];// raw f32 weight tile landed by TDM

  const int t    = threadIdx.x;
  const int lane = t & 31;
  const int g    = lane >> 4;
  const int lr   = lane & 15;
  const int wv   = t >> 5;
  const int wr   = (wv >> 2) * 32;
  const int wc   = (wv & 3) * 32;
  const int rowBase = blockIdx.y * 64;
  const int m0      = blockIdx.x * 128;

  // LDS byte offset of the TDM landing buffer
  const unsigned ldsoff =
      (unsigned)(uintptr_t)(__attribute__((address_space(3))) float*)lWraw;

  // TDM tile geometry (f32 elements):
  //   BT=false: tensor row len M, rows K; tile 128(cols) x 32(rows)
  //   BT=true : tensor row len K, rows M; tile 32(cols) x 128(rows)
  const unsigned tdim0  = (unsigned)(BT ? K : M);     // tensor_dim0 / stride
  const unsigned tdim1  = (unsigned)(BT ? M : K);
  const unsigned tile0  = BT ? 32u : 128u;            // tile_dim0 (row length)
  const unsigned tile1  = BT ? 128u : 32u;            // tile_dim1 (rows)

  v4i z4 = {};
  v8i z8 = {};
  (void)z8;

  v8f zero = {};
  v8f acc[2][2];
#pragma unroll
  for (int r = 0; r < 2; ++r)
#pragma unroll
    for (int c = 0; c < 2; ++c) acc[r][c] = zero;

  for (int k0 = 0; k0 < K; k0 += 32) {
    __syncthreads();   // previous iteration fully consumed

    // ---- issue TDM for the weight tile (wave 0 only; EXEC ignored) ----
    if (wv == 0) {
      const float* tile =
          W + (BT ? ((size_t)m0 * K + k0) : ((size_t)k0 * M + m0));
      uint64_t ga = (uint64_t)(uintptr_t)tile;
      v4u g0;
      g0[0] = 1u;                                  // count=1, user descriptor
      g0[1] = ldsoff;                              // lds_addr
      g0[2] = (unsigned)ga;                        // global_addr[31:0]
      g0[3] = (unsigned)((ga >> 32) & 0x01FFFFFFu) // global_addr[56:32]
            | (2u << 30);                          // type=2 (image)
      v8i g1;
      g1[0] = (int)(2u << 16);                     // data_size=4B, no mask
      g1[1] = (int)((tdim0 & 0xFFFFu) << 16);      // tensor_dim0[15:0]
      g1[2] = (int)((tdim0 >> 16) | ((tdim1 & 0xFFFFu) << 16));
      g1[3] = (int)((tdim1 >> 16) | (tile0 << 16));// tile_dim0
      g1[4] = (int)tile1;                          // tile_dim1 (tile_dim2=0)
      g1[5] = (int)tdim0;                          // tensor_dim0_stride[31:0]
      g1[6] = 0;                                   // stride hi / dim1_stride lo
      g1[7] = 0;
#if defined(__clang_major__) && __clang_major__ >= 23
      __builtin_amdgcn_tensor_load_to_lds(g0, g1, z4, z4, z8, 0);
#else
      __builtin_amdgcn_tensor_load_to_lds(g0, g1, z4, z4, 0);
#endif
    }

    // ---- stage A tile 64x32 f16 via regular vmem while TDM runs ----
    {
      int row = t >> 2;
      int col = (t & 3) * 8;
      const _Float16* src = A + (size_t)(rowBase + row) * K + (k0 + col);
#pragma unroll
      for (int j = 0; j < 8; ++j) lA[row][col + j] = src[j];
      if (k0 + 32 < K) __builtin_prefetch(src + 32, 0, 0);
    }

    if (wv == 0) __builtin_amdgcn_s_wait_tensorcnt(0);
    __syncthreads();   // raw weight tile visible to all waves

    // ---- convert raw f32 tile -> f16 transposed fragment layout ----
    {
      int m = t >> 1, kk0 = (t & 1) * 16;   // 128 cols x 2 halves of 16 k
      if (!BT) {
#pragma unroll
        for (int j = 0; j < 16; ++j)
          lBt[m][kk0 + j] = (_Float16)lWraw[(kk0 + j) * 128 + m];
      } else {
#pragma unroll
        for (int j = 0; j < 16; ++j)
          lBt[m][kk0 + j] = (_Float16)lWraw[m * 32 + kk0 + j];
      }
    }
    __syncthreads();

    // ---- fragments + WMMA ----
    v16h af[2], bf[2];
#pragma unroll
    for (int rt = 0; rt < 2; ++rt) {
      int m = wr + rt * 16 + lr;
#pragma unroll
      for (int e = 0; e < 16; ++e) {
        int kk = (e < 8) ? (g * 8 + e) : (16 + g * 8 + (e - 8));
        af[rt][e] = lA[m][kk];
      }
    }
#pragma unroll
    for (int ct = 0; ct < 2; ++ct) {
      int n = wc + ct * 16 + lr;
#pragma unroll
      for (int j = 0; j < 16; ++j) bf[ct][j] = lBt[n][(j < 8) ? (g * 8 + j) : (16 + g * 8 + (j - 8))];
    }
#pragma unroll
    for (int rt = 0; rt < 2; ++rt)
#pragma unroll
      for (int ct = 0; ct < 2; ++ct)
        acc[rt][ct] = __builtin_amdgcn_wmma_f32_16x16x32_f16(
            false, af[rt], false, bf[ct], (short)0, acc[rt][ct], false, false);
  }

  // ---- branch-free epilogue (layout: element i -> row 8g+i, col l&15) ----
#pragma unroll
  for (int ct = 0; ct < 2; ++ct) {
    const int gn = m0 + wc + ct * 16 + lr;
    const float bv = HB ? bias[gn] : 0.0f;
#pragma unroll
    for (int rt = 0; rt < 2; ++rt)
#pragma unroll
      for (int i = 0; i < 8; ++i) {
        const int gm = rowBase + wr + rt * 16 + g * 8 + i;
        float v = acc[rt][ct][i] + bv;
        if (HR)   v += resid[(size_t)gm * M + gn];
        if (RELU) v = fmaxf(v, 0.0f);
        if (OF) outF[(size_t)gm * M + gn] = v;
        if (OH) outH[(size_t)gm * M + gn] = (_Float16)v;
      }
  }
}

// ---------------------------------------------------------------------------
// Flash attention: one wave per (batch, head, 16-query tile). S=1024, B=4.
// D = head dim (32 for 8 heads, 64 for 4 heads). Token row n = s*4 + b.
// ---------------------------------------------------------------------------
template<int D>
__global__ __launch_bounds__(128) void flash_attn(
    const _Float16* __restrict__ Qb, int ldq,
    const _Float16* __restrict__ Kb, int ldk,
    const _Float16* __restrict__ Vb, int ldv,
    _Float16* __restrict__ Ob)
{
  constexpr int HEADS = 256 / D;
  __shared__ _Float16 lP[4][16][40];
  const int t = threadIdx.x;
  const int lane = t & 31, g = lane >> 4, lr = lane & 15;
  const int wv = t >> 5;
  const int w  = blockIdx.x * 4 + wv;
  const int qt = w & 63;
  const int h  = (w >> 6) % HEADS;
  const int b  = (w >> 6) / HEADS;

  v8f zero = {};
  v16h qf[D / 32];
#pragma unroll
  for (int dsi = 0; dsi < D / 32; ++dsi) {
    int s = qt * 16 + lr;
    size_t base = ((size_t)s * 4 + b) * ldq + h * D + dsi * 32;
#pragma unroll
    for (int e = 0; e < 16; ++e) {
      int kk = (e < 8) ? (g * 8 + e) : (16 + g * 8 + (e - 8));
      qf[dsi][e] = Qb[base + kk];
    }
  }
  float mrun[8], lrun[8];
#pragma unroll
  for (int i = 0; i < 8; ++i) { mrun[i] = -3.0e38f; lrun[i] = 0.0f; }
  v8f oacc[D / 16];
#pragma unroll
  for (int nt = 0; nt < D / 16; ++nt) oacc[nt] = zero;
  const float scale = rsqrtf((float)D);

  for (int kb = 0; kb < 32; ++kb) {   // 32 keys per iteration
    v8f sf[2]; sf[0] = zero; sf[1] = zero;
#pragma unroll
    for (int ct = 0; ct < 2; ++ct) {
      int key = kb * 32 + ct * 16 + lr;
#pragma unroll
      for (int dsi = 0; dsi < D / 32; ++dsi) {
        v16h kf;
        size_t kbase = ((size_t)key * 4 + b) * ldk + h * D + dsi * 32;
#pragma unroll
        for (int j = 0; j < 16; ++j) kf[j] = Kb[kbase + ((j < 8) ? (g * 8 + j) : (16 + g * 8 + (j - 8)))];
        sf[ct] = __builtin_amdgcn_wmma_f32_16x16x32_f16(
            false, qf[dsi], false, kf, (short)0, sf[ct], false, false);
      }
    }
    // Online softmax: row of element i is (8g+i); 16 cols live across lanes
#pragma unroll
    for (int i = 0; i < 8; ++i) {
      float s0 = sf[0][i] * scale;
      float s1 = sf[1][i] * scale;
      float x = fmaxf(s0, s1);
#pragma unroll
      for (int off = 1; off < 16; off <<= 1) x = fmaxf(x, __shfl_xor(x, off, 32));
      float mnew = fmaxf(mrun[i], x);
      float corr = __expf(mrun[i] - mnew);
      float p0 = __expf(s0 - mnew);
      float p1 = __expf(s1 - mnew);
      float ps = p0 + p1;
#pragma unroll
      for (int off = 1; off < 16; off <<= 1) ps += __shfl_xor(ps, off, 32);
      lrun[i] = lrun[i] * corr + ps;
      mrun[i] = mnew;
#pragma unroll
      for (int nt = 0; nt < D / 16; ++nt) oacc[nt][i] *= corr;
      int m = g * 8 + i;
      lP[wv][m][lr]      = (_Float16)p0;
      lP[wv][m][lr + 16] = (_Float16)p1;
    }
    __builtin_amdgcn_wave_barrier();
    asm volatile("s_wait_dscnt 0" ::: "memory");
    v16h pf;  // reshape P [16q,32keys] into an A fragment
#pragma unroll
    for (int e = 0; e < 16; ++e) {
      int kk = (e < 8) ? (g * 8 + e) : (16 + g * 8 + (e - 8));
      pf[e] = lP[wv][lr][kk];
    }
    __builtin_amdgcn_wave_barrier();
#pragma unroll
    for (int nt = 0; nt < D / 16; ++nt) {
      v16h vf;
#pragma unroll
      for (int j = 0; j < 16; ++j) {
        int key = kb * 32 + ((j < 8) ? (g * 8 + j) : (16 + g * 8 + (j - 8)));
        vf[j] = Vb[((size_t)key * 4 + b) * ldv + h * D + nt * 16 + lr];
      }
      oacc[nt] = __builtin_amdgcn_wmma_f32_16x16x32_f16(
          false, pf, false, vf, (short)0, oacc[nt], false, false);
    }
  }
#pragma unroll
  for (int i = 0; i < 8; ++i) lrun[i] = 1.0f / lrun[i];
#pragma unroll
  for (int nt = 0; nt < D / 16; ++nt)
#pragma unroll
    for (int i = 0; i < 8; ++i) {
      int s = qt * 16 + g * 8 + i;
      int col = h * D + nt * 16 + lr;
      Ob[((size_t)s * 4 + b) * 256 + col] = (_Float16)(oacc[nt][i] * lrun[i]);
    }
}

// ---------------------------------------------------------------------------
// Elementwise / reduction kernels
// ---------------------------------------------------------------------------
__device__ __forceinline__ float block_reduce_sum(float v, float* sbuf) {
#pragma unroll
  for (int o = 16; o > 0; o >>= 1) v += __shfl_xor(v, o, 32);
  int lane = threadIdx.x & 31, wv = threadIdx.x >> 5;
  if (lane == 0) sbuf[wv] = v;
  __syncthreads();
  float r = 0.0f;
  if (threadIdx.x < 8) r = sbuf[threadIdx.x];
  if (threadIdx.x < 32) {
#pragma unroll
    for (int o = 4; o > 0; o >>= 1) r += __shfl_xor(r, o, 32);
    if (threadIdx.x == 0) sbuf[0] = r;
  }
  __syncthreads();
  r = sbuf[0];
  __syncthreads();
  return r;
}

// x [B,E,H,W] -> xseq [n=s*4+b, e] (f32 + f16)
__global__ void pack_x(const float* __restrict__ x,
                       float* __restrict__ xF, _Float16* __restrict__ xH) {
  int idx = blockIdx.x * 256 + threadIdx.x;      // B*E*S = 1,048,576
  int s = idx & 1023;
  int e = (idx >> 10) & 255;
  int b = idx >> 18;
  float v = x[idx];
  size_t o = (size_t)(s * 4 + b) * 256 + e;
  xF[o] = v;
  xH[o] = (_Float16)v;
}

// ref[k] = (sum_h - h[k]) / 2 ; also make f16 copies of h
__global__ void ref_kernel(const float* __restrict__ h0, const float* __restrict__ h1,
                           const float* __restrict__ h2,
                           _Float16* H0, _Float16* H1, _Float16* H2,
                           _Float16* r0, _Float16* r1, _Float16* r2) {
  int i = blockIdx.x * 256 + threadIdx.x;        // NTOK*EDIM
  float a = h0[i], b = h1[i], c = h2[i];
  float s = a + b + c;
  H0[i] = (_Float16)a; H1[i] = (_Float16)b; H2[i] = (_Float16)c;
  r0[i] = (_Float16)((s - a) * 0.5f);
  r1[i] = (_Float16)((s - b) * 0.5f);
  r2[i] = (_Float16)((s - c) * 0.5f);
}

// refined = LN(h + cf); writes d_out slot nh ([B,E,S] layout) and flat16
__global__ __launch_bounds__(256) void ln_refined(
    const float* __restrict__ H, const float* __restrict__ CF,
    const float* __restrict__ gam, const float* __restrict__ bet,
    _Float16* __restrict__ flat, float* __restrict__ out, int nh) {
  __shared__ float sbuf[32];
  int n = blockIdx.x, e = threadIdx.x;
  float v = H[(size_t)n * 256 + e] + CF[(size_t)n * 256 + e];
  float mean = block_reduce_sum(v, sbuf) * (1.0f / 256.0f);
  float d = v - mean;
  float var = block_reduce_sum(d * d, sbuf) * (1.0f / 256.0f);
  float y = d * rsqrtf(var + 1e-5f) * gam[e] + bet[e];
  flat[(size_t)n * 768 + nh * 256 + e] = (_Float16)y;
  int b = n & 3, s = n >> 2;
  out[((size_t)nh * 4 + b) * (256 * 1024) + (size_t)e * 1024 + s] = y;
}

// fusion logits (256->3) + softmax, one thread per token
__global__ void fus_weights(const _Float16* __restrict__ f1,
                            const float* __restrict__ w2, const float* __restrict__ b2,
                            float* __restrict__ wout) {
  int n = blockIdx.x * 256 + threadIdx.x;
  float l0 = b2[0], l1 = b2[1], l2 = b2[2];
  for (int j = 0; j < 256; ++j) {
    float a = (float)f1[(size_t)n * 256 + j];
    l0 += a * w2[j * 3 + 0];
    l1 += a * w2[j * 3 + 1];
    l2 += a * w2[j * 3 + 2];
  }
  float mx = fmaxf(l0, fmaxf(l1, l2));
  float e0 = __expf(l0 - mx), e1 = __expf(l1 - mx), e2 = __expf(l2 - mx);
  float inv = 1.0f / (e0 + e1 + e2);
  wout[n * 3 + 0] = e0 * inv;
  wout[n * 3 + 1] = e1 * inv;
  wout[n * 3 + 2] = e2 * inv;
}

// fused = LN1(sum_k w_k*refined_k + x); ln2H = LN2(fused) as f16
__global__ __launch_bounds__(256) void fuse_ln(
    const float* __restrict__ w, const _Float16* __restrict__ flat,
    const float* __restrict__ xseq,
    const float* __restrict__ g1, const float* __restrict__ b1,
    const float* __restrict__ g2, const float* __restrict__ b2,
    float* __restrict__ fusedF, _Float16* __restrict__ ln2H) {
  __shared__ float sbuf[32];
  int n = blockIdx.x, e = threadIdx.x;
  float acc = 0.0f;
#pragma unroll
  for (int k = 0; k < 3; ++k)
    acc += w[n * 3 + k] * (float)flat[(size_t)n * 768 + k * 256 + e];
  float v = acc + xseq[(size_t)n * 256 + e];
  float mean = block_reduce_sum(v, sbuf) * (1.0f / 256.0f);
  float d = v - mean;
  float var = block_reduce_sum(d * d, sbuf) * (1.0f / 256.0f);
  float y = d * rsqrtf(var + 1e-5f) * g1[e] + b1[e];
  fusedF[(size_t)n * 256 + e] = y;
  float mean2 = block_reduce_sum(y, sbuf) * (1.0f / 256.0f);
  float d2 = y - mean2;
  float var2 = block_reduce_sum(d2 * d2, sbuf) * (1.0f / 256.0f);
  ln2H[(size_t)n * 256 + e] = (_Float16)(d2 * rsqrtf(var2 + 1e-5f) * g2[e] + b2[e]);
}

// out slot 3 = LN3(fused + ff)
__global__ __launch_bounds__(256) void final_ln(
    const float* __restrict__ fusedF, const float* __restrict__ ff,
    const float* __restrict__ g3, const float* __restrict__ b3,
    float* __restrict__ out) {
  __shared__ float sbuf[32];
  int n = blockIdx.x, e = threadIdx.x;
  float v = fusedF[(size_t)n * 256 + e] + ff[(size_t)n * 256 + e];
  float mean = block_reduce_sum(v, sbuf) * (1.0f / 256.0f);
  float d = v - mean;
  float var = block_reduce_sum(d * d, sbuf) * (1.0f / 256.0f);
  float y = d * rsqrtf(var + 1e-5f) * g3[e] + b3[e];
  int b = n & 3, s = n >> 2;
  out[((size_t)3 * 4 + b) * (256 * 1024) + (size_t)e * 1024 + s] = y;
}

// ---------------------------------------------------------------------------
extern "C" void kernel_launch(void* const* d_in, const int* in_sizes, int n_in,
                              void* d_out, int out_size, void* d_ws, size_t ws_size,
                              hipStream_t stream) {
  (void)in_sizes; (void)n_in; (void)out_size; (void)ws_size;
  const int N = NTOK, E = EDIM, FF = FDIM;

  const float* x         = (const float*)d_in[0];
  const float* int_w1    = (const float*)d_in[1];
  const float* int_b1    = (const float*)d_in[2];
  const float* int_w2    = (const float*)d_in[3];
  const float* int_b2    = (const float*)d_in[4];
  const float* hyp_in_w  = (const float*)d_in[5];
  const float* hyp_in_b  = (const float*)d_in[6];
  const float* hyp_out_w = (const float*)d_in[7];
  const float* hyp_out_b = (const float*)d_in[8];
  const float* cf_in_w   = (const float*)d_in[9];
  const float* cf_in_b   = (const float*)d_in[10];
  const float* cf_out_w  = (const float*)d_in[11];
  const float* cf_out_b  = (const float*)d_in[12];
  const float* dis_w1    = (const float*)d_in[13];
  const float* dis_b1    = (const float*)d_in[14];
  const float* dis_w2    = (const float*)d_in[15];
  const float* dis_b2    = (const float*)d_in[16];
  const float* ffn_w1    = (const float*)d_in[17];
  const float* ffn_b1    = (const float*)d_in[18];
  const float* ffn_w2    = (const float*)d_in[19];
  const float* ffn_b2    = (const float*)d_in[20];
  const float* norm_g    = (const float*)d_in[21];
  const float* norm_b    = (const float*)d_in[22];
  const float* fus_w1    = (const float*)d_in[23];
  const float* fus_b1    = (const float*)d_in[24];
  const float* fus_w2    = (const float*)d_in[25];
  const float* fus_b2    = (const float*)d_in[26];
  float* out = (float*)d_out;

  // ---- workspace carve-out (all offsets 256B aligned) ----
  char* ws = (char*)d_ws;
  size_t off = 0;
  auto alloc = [&](size_t bytes) -> char* {
    char* p = ws + off;
    off += (bytes + 255) & ~(size_t)255;
    return p;
  };
  float*    xseqF   = (float*)   alloc((size_t)N * E * 4);
  _Float16* xseqH   = (_Float16*)alloc((size_t)N * E * 2);
  float*    causalF = (float*)   alloc((size_t)N * E * 4);
  _Float16* causalH = (_Float16*)alloc((size_t)N * E * 2);
  _Float16* buf512H = (_Float16*)alloc((size_t)N * FF * 2);   // h1 / ffn hidden
  _Float16* t128H   = (_Float16*)alloc((size_t)N * 128 * 2);
  _Float16* ivH     = (_Float16*)alloc((size_t)N * E * 2);
  _Float16* qkvH    = (_Float16*)alloc((size_t)N * 768 * 2);  // also cfq|cfkv
  _Float16* attnH   = (_Float16*)alloc((size_t)N * E * 2);
  float*    hF[3]; _Float16* hH[3]; _Float16* refH[3];
  for (int k = 0; k < 3; ++k) hF[k]   = (float*)   alloc((size_t)N * E * 4);
  for (int k = 0; k < 3; ++k) hH[k]   = (_Float16*)alloc((size_t)N * E * 2);
  for (int k = 0; k < 3; ++k) refH[k] = (_Float16*)alloc((size_t)N * E * 2);
  float*    cfF     = (float*)   alloc((size_t)N * E * 4);
  _Float16* flatH   = (_Float16*)alloc((size_t)N * 768 * 2);
  float*    wF      = (float*)   alloc((size_t)N * 3 * 4);
  _Float16* f1H     = (_Float16*)alloc((size_t)N * E * 2);
  float*    fusedF  = (float*)   alloc((size_t)N * E * 4);
  _Float16* ln2H    = (_Float16*)alloc((size_t)N * E * 2);
  float*    ffF     = (float*)   alloc((size_t)N * E * 4);

  const dim3 blk(256);
  const int GY = N / 64;  // 64 row-tiles

  // 0) [B,E,H,W] -> token-major sequence
  pack_x<<<4096, blk, 0, stream>>>(x, xseqF, xseqH);

  // 1) causal = MLP(x): 256 -> 512 (relu) -> 256
  gemm_f16_wmma<false, true , true, false, false, true>
      <<<dim3(FF / 128, GY), blk, 0, stream>>>(
      xseqH, dis_w1, dis_b1, nullptr, nullptr, buf512H, E, FF);
  gemm_f16_wmma<false, false, true, false, true , true>
      <<<dim3(E / 128, GY), blk, 0, stream>>>(
      buf512H, dis_w2, dis_b2, nullptr, causalF, causalH, FF, E);

  // 2) per-hypothesis: intervention MLP + self-attention (8 heads, d=32)
  for (int nh = 0; nh < 3; ++nh) {
    gemm_f16_wmma<false, true , true, false, false, true>
        <<<dim3(1, GY), blk, 0, stream>>>(
        causalH, int_w1 + (size_t)nh * E * 128, int_b1 + nh * 128,
        nullptr, nullptr, t128H, E, 128);
    gemm_f16_wmma<false, false, true, true , false, true>
        <<<dim3(E / 128, GY), blk, 0, stream>>>(
        t128H, int_w2 + (size_t)nh * 128 * E, int_b2 + nh * E,
        causalF, nullptr, ivH, 128, E);
    gemm_f16_wmma<true , false, true, false, false, true>
        <<<dim3(768 / 128, GY), blk, 0, stream>>>(
        ivH, hyp_in_w + (size_t)nh * 768 * E, hyp_in_b + nh * 768,
        nullptr, nullptr, qkvH, E, 768);
    flash_attn<32><<<512, 128, 0, stream>>>(qkvH, 768, qkvH + E, 768,
                                            qkvH + 2 * E, 768, attnH);
    gemm_f16_wmma<true , false, true, false, true , false>
        <<<dim3(E / 128, GY), blk, 0, stream>>>(
        attnH, hyp_out_w + (size_t)nh * E * E, hyp_out_b + nh * E,
        nullptr, hF[nh], nullptr, E, E);
  }

  // 3) reference hypotheses (mean of the others) + f16 copies
  ref_kernel<<<4096, blk, 0, stream>>>(hF[0], hF[1], hF[2],
                                       hH[0], hH[1], hH[2],
                                       refH[0], refH[1], refH[2]);

  // 4) counterfactual attention (4 heads, d=64) + LN -> refined outputs
  _Float16* cfq  = qkvH;
  _Float16* cfkv = qkvH + (size_t)N * E;
  for (int nh = 0; nh < 3; ++nh) {
    gemm_f16_wmma<true , false, true, false, false, true>
        <<<dim3(E / 128, GY), blk, 0, stream>>>(
        hH[nh], cf_in_w + (size_t)nh * 768 * E, cf_in_b + nh * 768,
        nullptr, nullptr, cfq, E, E);
    gemm_f16_wmma<true , false, true, false, false, true>
        <<<dim3(512 / 128, GY), blk, 0, stream>>>(
        refH[nh], cf_in_w + (size_t)nh * 768 * E + (size_t)E * E,
        cf_in_b + nh * 768 + E, nullptr, nullptr, cfkv, E, 512);
    flash_attn<64><<<256, 128, 0, stream>>>(cfq, 256, cfkv, 512,
                                            cfkv + E, 512, attnH);
    gemm_f16_wmma<true , false, true, false, true , false>
        <<<dim3(E / 128, GY), blk, 0, stream>>>(
        attnH, cf_out_w + (size_t)nh * E * E, cf_out_b + nh * E,
        nullptr, cfF, nullptr, E, E);
    ln_refined<<<N, blk, 0, stream>>>(hF[nh], cfF, norm_g, norm_b,
                                      flatH, out, nh);
  }

  // 5) fusion: weights = softmax(relu(flat@W1+b1)@W2+b2); fused = LN1(sum+x)
  gemm_f16_wmma<false, true , true, false, false, true>
      <<<dim3(E / 128, GY), blk, 0, stream>>>(
      flatH, fus_w1, fus_b1, nullptr, nullptr, f1H, 768, E);
  fus_weights<<<N / 256, blk, 0, stream>>>(f1H, fus_w2, fus_b2, wF);
  fuse_ln<<<N, blk, 0, stream>>>(wF, flatH, xseqF,
                                 norm_g + E, norm_b + E,
                                 norm_g + 2 * E, norm_b + 2 * E,
                                 fusedF, ln2H);

  // 6) FFN + final LN -> output slot 3
  gemm_f16_wmma<false, true , true, false, false, true>
      <<<dim3(FF / 128, GY), blk, 0, stream>>>(
      ln2H, ffn_w1, ffn_b1, nullptr, nullptr, buf512H, E, FF);
  gemm_f16_wmma<false, false, true, false, true , false>
      <<<dim3(E / 128, GY), blk, 0, stream>>>(
      buf512H, ffn_w2, ffn_b2, nullptr, ffF, nullptr, FF, E);
  final_ln<<<N, blk, 0, stream>>>(fusedF, ffF, norm_g + 3 * E, norm_b + 3 * E, out);
}
